// MH_SelfAttentionWide_73194832658893
// MI455X (gfx1250) — compile-verified, matrix operand order checked by reading, not run on hardware
//
#include <hip/hip_runtime.h>
#include <hip/hip_bf16.h>

// ---------------------------------------------------------------------------
// MI455X (gfx1250) wave32 WMMA implementation of wide multi-head attention.
//   B=2, S=2048, D=512 (per head), H=8
// All heavy math goes through v_wmma_f32_16x16x32_bf16 (f32 accumulate).
// ---------------------------------------------------------------------------

#define B_  2
#define S_  2048
#define D_  512
#define H_  8
#define HD_ (H_ * D_)

// LDS row paddings chosen so the 16 lanes of a fragment load hit disjoint
// 4-bank spans (row strides of 20 dwords, or 260 dwords for the K tile).
#define KT_PITCH 520   // halves per K-tile row   (1040 B, 16B aligned)
#define VT_PITCH 40    // halves per V-tile row   (  80 B, 16B aligned)
#define PB_PITCH 40    // halves per P-tile row
#define AS_PITCH 40    // GEMM A tile row pitch
#define BS_PITCH 40    // GEMM B tile row pitch

typedef __attribute__((ext_vector_type(16))) __bf16          v16bf;
typedef __attribute__((ext_vector_type(8)))  float           v8f;
typedef __attribute__((ext_vector_type(16))) unsigned short  v16us;
typedef __attribute__((ext_vector_type(8)))  unsigned short  v8us;

__device__ __forceinline__ unsigned short f2bf(float f) {
    unsigned int u = __builtin_bit_cast(unsigned int, f);
    u += 0x7FFFu + ((u >> 16) & 1u);      // round-to-nearest-even
    return (unsigned short)(u >> 16);
}

// Build a 16x32 bf16 A/B fragment per CDNA5 layout:
// lane L holds row/col (L&15); lanes<16 carry K={0..7,16..23}, lanes>=16
// carry K={8..15,24..31}; caller passes p = rowBase + (lane>>4)*8, so we
// read 8 halves at p and 8 halves at p+16 (two 16-byte loads).
__device__ __forceinline__ v16bf ldfrag(const unsigned short* p) {
    v8us a = *(const v8us*)p;
    v8us b = *(const v8us*)(p + 16);
    v16us u = __builtin_shufflevector(a, b, 0,1,2,3,4,5,6,7,8,9,10,11,12,13,14,15);
    return __builtin_bit_cast(v16bf, u);
}

__device__ __forceinline__ v8f wmma_bf16(v16bf a, v16bf b, v8f c) {
    return __builtin_amdgcn_wmma_f32_16x16x32_bf16(false, a, false, b,
                                                   (short)0, c, false, false);
}

// ---------------------------------------------------------------------------
// Zero-fill (attn_avg is accumulated with atomics)
// ---------------------------------------------------------------------------
__global__ void zero_f32(float* __restrict__ p, size_t n) {
    size_t i = (size_t)blockIdx.x * blockDim.x + threadIdx.x;
    if (i < n) p[i] = 0.0f;
}

// ---------------------------------------------------------------------------
// Generic WMMA GEMM:  C[z] = scale * A[z] @ B[z] (+ bias)
//   A: [M,K] row-major (f32 or bf16), B: [K,N] row-major (f32 or bf16)
//   C: [M,N] (bf16 or f32). Batch offsets: aOff=(z/aDiv)*aStride,
//   bOff=(z%bMod)*bStride, cOff=z*cStride.
// Block = 256 threads = 8 waves; macro tile 128(M) x 64(N); K-step 32.
// Each wave computes a 32x32 region = 2x2 WMMA tiles.
// ---------------------------------------------------------------------------
template<bool A_F32, bool B_F32, bool OUT_BF16, bool BIAS>
__global__ __launch_bounds__(256)
void gemm_wmma(const void* __restrict__ Ap, const void* __restrict__ Bp,
               void* __restrict__ Cp, const float* __restrict__ bias,
               int M, int N, int K,
               long long aStride, int aDiv, long long bStride, int bMod,
               long long cStride, float scale)
{
    __shared__ __align__(16) unsigned short As[128][AS_PITCH];  // (m,k)
    __shared__ __align__(16) unsigned short Bs[64][BS_PITCH];   // (n,k) transposed

    const int z    = blockIdx.z;
    const int tid  = threadIdx.x;
    const int lane = tid & 31;
    const int w    = tid >> 5;
    const int m0   = blockIdx.y * 128;
    const int n0   = blockIdx.x * 64;
    const int wm   = (w & 3) * 32;   // wave row base within macro tile
    const int wn   = (w >> 2) * 32;  // wave col base within macro tile

    const size_t aOff = (size_t)(z / aDiv) * (size_t)aStride;
    const size_t bOff = (size_t)(z % bMod) * (size_t)bStride;
    const size_t cOff = (size_t)z * (size_t)cStride;

    // staging coordinates
    const int ar = tid >> 1;            // A row 0..127
    const int ak = (tid & 1) * 16;      // A k base {0,16}
    const int bk = tid >> 3;            // B k row 0..31
    const int bn = (tid & 7) * 8;       // B col base 0..56

    // fragment coordinates
    const int fr = lane & 15;
    const int fk = (lane >> 4) * 8;
    const int cm = (lane >> 4) * 8;     // C: vgpr r -> m = r + cm
    const int cn = lane & 15;

    v8f acc00 = {}, acc01 = {}, acc10 = {}, acc11 = {};

    for (int k0 = 0; k0 < K; k0 += 32) {
        // ---- stage A tile (convert f32 -> bf16 on the fly) ----
        if (A_F32) {
            const float* A = (const float*)Ap + aOff + (size_t)(m0 + ar) * K + k0 + ak;
            #pragma unroll
            for (int i = 0; i < 4; ++i) {
                float4 v = *(const float4*)(A + 4 * i);
                As[ar][ak + 4*i + 0] = f2bf(v.x);
                As[ar][ak + 4*i + 1] = f2bf(v.y);
                As[ar][ak + 4*i + 2] = f2bf(v.z);
                As[ar][ak + 4*i + 3] = f2bf(v.w);
            }
        } else {
            const unsigned short* A =
                (const unsigned short*)Ap + aOff + (size_t)(m0 + ar) * K + k0 + ak;
            *(v8us*)&As[ar][ak]     = *(const v8us*)(A);
            *(v8us*)&As[ar][ak + 8] = *(const v8us*)(A + 8);
        }
        // ---- stage B tile transposed (coalesced read, scalar LDS write) ----
        if (B_F32) {
            const float* Bm = (const float*)Bp + bOff + (size_t)(k0 + bk) * N + n0 + bn;
            #pragma unroll
            for (int i = 0; i < 8; ++i) Bs[bn + i][bk] = f2bf(Bm[i]);
        } else {
            const unsigned short* Bm =
                (const unsigned short*)Bp + bOff + (size_t)(k0 + bk) * N + n0 + bn;
            #pragma unroll
            for (int i = 0; i < 8; ++i) Bs[bn + i][bk] = Bm[i];
        }
        __syncthreads();

        v16bf a0 = ldfrag(&As[wm      + fr][fk]);
        v16bf a1 = ldfrag(&As[wm + 16 + fr][fk]);
        v16bf b0 = ldfrag(&Bs[wn      + fr][fk]);
        v16bf b1 = ldfrag(&Bs[wn + 16 + fr][fk]);

        acc00 = wmma_bf16(a0, b0, acc00);
        acc01 = wmma_bf16(a0, b1, acc01);
        acc10 = wmma_bf16(a1, b0, acc10);
        acc11 = wmma_bf16(a1, b1, acc11);
        __syncthreads();
    }

    // ---- epilogue: C layout = vgpr r: m=r+8*(lane>>4), n=lane&15 ----
    #pragma unroll
    for (int mt = 0; mt < 2; ++mt) {
        #pragma unroll
        for (int nt = 0; nt < 2; ++nt) {
            v8f acc = (mt == 0) ? (nt == 0 ? acc00 : acc01)
                                : (nt == 0 ? acc10 : acc11);
            const int col = n0 + wn + nt * 16 + cn;
            #pragma unroll
            for (int r = 0; r < 8; ++r) {
                const int m = m0 + wm + mt * 16 + cm + r;
                float v = acc[r] * scale;
                if (BIAS) v += bias[col];
                const size_t idx = cOff + (size_t)m * N + col;
                if (OUT_BF16) ((unsigned short*)Cp)[idx] = f2bf(v);
                else          ((float*)Cp)[idx] = v;
            }
        }
    }
}

// ---------------------------------------------------------------------------
// Flash-style attention per (b, h, 64 q-rows). Two passes over KV:
//   pass 1: exact running max/sum per q row (no attn materialization)
//   pass 2: p = exp(s-m)/l -> bf16 P tile; atomicAdd p/H into attn_avg;
//           O += P @ V via WMMA. O kept in 16x v8f per wave (128 VGPRs).
// Q fragments are register-resident (hoisted by the compiler). K and V tiles
// are staged cooperatively in a shared LDS buffer (never live at the same
// time), so per-kt fragment loads are short-latency ds_load_b128 instead of
// duplicated L2-latency global loads.
// ---------------------------------------------------------------------------
__global__ __launch_bounds__(256)
void attn_kernel(const unsigned short* __restrict__ Q,
                 const unsigned short* __restrict__ K,
                 const unsigned short* __restrict__ V,
                 unsigned short* __restrict__ Zc,
                 float* __restrict__ attn_avg)
{
    // K tile: [32][KT_PITCH] row-major (kv, d)   -> 16640 halves
    // V tile: [512][VT_PITCH] transposed (d, kv) -> 20480 halves
    __shared__ __align__(16) unsigned short KVbuf[512 * VT_PITCH]; // 40 KB
    __shared__ float          Sb[64][33];                          // 8.25 KB
    __shared__ __align__(16) unsigned short Pb[64][PB_PITCH];      //  5 KB
    __shared__ float mrow[64], lrow[64];

    const int b  = blockIdx.z;
    const int h  = blockIdx.y;
    const int q0 = blockIdx.x * 64;

    const int tid  = threadIdx.x;
    const int lane = tid & 31;
    const int w    = tid >> 5;
    const int mt   = w & 3;          // q-tile (16 rows) of this wave
    const int nt   = w >> 2;         // kv half / output col half

    const size_t base = ((size_t)(b * H_ + h)) * S_ * D_;
    const unsigned short* Qh = Q + base;
    const unsigned short* Kh = K + base;
    const unsigned short* Vh = V + base;

    const int fr = lane & 15;
    const int fk = (lane >> 4) * 8;

    // cooperative K-tile staging: straight copy, vector ds stores
    const int sr = tid >> 3;         // kv row 0..31
    const int sd = (tid & 7) * 64;   // d base
    auto stage_k = [&](int kv0) {
        const unsigned short* src = Kh + (size_t)(kv0 + sr) * D_ + sd;
        #pragma unroll
        for (int i = 0; i < 8; ++i)
            *(v8us*)&KVbuf[sr * KT_PITCH + sd + i * 8] = *(const v8us*)(src + i * 8);
    };

    // scores for one 64x32 tile: wave w computes 16x16 tile (mt, nt),
    // K fragments from LDS, dual accumulators to break the RAW chain.
    auto compute_scores = [&](int kv0) {
        const unsigned short* qrow = Qh + (size_t)(q0 + mt * 16 + fr) * D_ + fk;
        const unsigned short* krow = &KVbuf[(nt * 16 + fr) * KT_PITCH + fk];
        v8f s0 = {}, s1 = {};
        #pragma unroll
        for (int kt = 0; kt < D_ / 32; ++kt) {
            v16bf a  = ldfrag(qrow + kt * 32);
            v16bf bb = ldfrag(krow + kt * 32);
            if (kt & 1) s1 = wmma_bf16(a, bb, s1);
            else        s0 = wmma_bf16(a, bb, s0);
        }
        #pragma unroll
        for (int r = 0; r < 8; ++r)
            Sb[mt * 16 + (lane >> 4) * 8 + r][nt * 16 + fr] = s0[r] + s1[r];
    };

    // ---------------- pass 1: softmax statistics ----------------
    if (tid < 64) { mrow[tid] = -1e30f; lrow[tid] = 0.0f; }
    __syncthreads();
    for (int kv0 = 0; kv0 < S_; kv0 += 32) {
        stage_k(kv0);
        __syncthreads();
        compute_scores(kv0);
        __syncthreads();
        // stats (thr 0..63) overlap the next iteration's K staging
        if (tid < 64) {
            float m_old = mrow[tid], l = lrow[tid];
            float mx = m_old;
            #pragma unroll
            for (int c = 0; c < 32; ++c) mx = fmaxf(mx, Sb[tid][c]);
            float sum = 0.0f;
            #pragma unroll
            for (int c = 0; c < 32; ++c) sum += __expf(Sb[tid][c] - mx);
            lrow[tid] = l * __expf(m_old - mx) + sum;
            mrow[tid] = mx;
        }
    }
    __syncthreads();

    // ---------------- pass 2: probabilities + P@V ----------------
    v8f o[16];
    #pragma unroll
    for (int i = 0; i < 16; ++i) o[i] = (v8f){};
    const int nbase = nt * 256;      // this wave covers d in [nbase, nbase+256)

    const int pr = tid >> 2;         // prob step: q row
    const int pc = (tid & 3) * 8;    // prob step: kv col base

    for (int kv0 = 0; kv0 < S_; kv0 += 32) {
        stage_k(kv0);
        __syncthreads();
        compute_scores(kv0);
        __syncthreads();

        // stage V tile transposed Vt[d][kv] (overwrites K tile — scores done)
        {
            const unsigned short* src = Vh + (size_t)(kv0 + sr) * D_ + sd;
            #pragma unroll 8
            for (int i = 0; i < 64; ++i) KVbuf[(sd + i) * VT_PITCH + sr] = src[i];
        }
        // probabilities: exact (m,l known) -> Pb bf16 + attn_avg atomics
        {
            const float mr   = mrow[pr];
            const float invl = 1.0f / lrow[pr];
            float* avg = attn_avg + ((size_t)b * S_ + (q0 + pr)) * S_ + kv0 + pc;
            #pragma unroll
            for (int i = 0; i < 8; ++i) {
                float p = __expf(Sb[pr][pc + i] - mr) * invl;
                Pb[pr][pc + i] = f2bf(p);
                atomicAdd(avg + i, p * (1.0f / H_));
            }
        }
        __syncthreads();

        // O += P @ V : A = Pb (64x32), B = Vt (32 x 512), this wave: 16 tiles
        {
            v16bf a = ldfrag(&Pb[mt * 16 + fr][fk]);
            #pragma unroll
            for (int t = 0; t < 16; ++t) {
                v16bf bb = ldfrag(&KVbuf[(nbase + t * 16 + fr) * VT_PITCH + fk]);
                o[t] = wmma_bf16(a, bb, o[t]);
            }
        }
        __syncthreads();
    }

    // write O (bf16) into concatenated-head layout Zc[b, s, h*D + d]
    const int cm = (lane >> 4) * 8;
    const int cn = lane & 15;
    #pragma unroll
    for (int t = 0; t < 16; ++t) {
        #pragma unroll
        for (int r = 0; r < 8; ++r) {
            const int m = q0 + mt * 16 + cm + r;
            const int d = nbase + t * 16 + cn;
            Zc[((size_t)b * S_ + m) * HD_ + (size_t)h * D_ + d] = f2bf(o[t][r]);
        }
    }
}

// ---------------------------------------------------------------------------
// Host launcher
// ---------------------------------------------------------------------------
extern "C" void kernel_launch(void* const* d_in, const int* in_sizes, int n_in,
                              void* d_out, int out_size, void* d_ws, size_t ws_size,
                              hipStream_t stream)
{
    (void)in_sizes; (void)n_in; (void)out_size; (void)ws_size;

    const float* Xq = (const float*)d_in[0];
    const float* Xk = (const float*)d_in[1];
    const float* Xv = (const float*)d_in[2];
    const float* Wq = (const float*)d_in[3];
    const float* Wk = (const float*)d_in[4];
    const float* Wv = (const float*)d_in[5];
    const float* Wz = (const float*)d_in[6];
    const float* bz = (const float*)d_in[7];

    float* out      = (float*)d_out;                       // [B,S,D]
    float* attn_avg = out + (size_t)B_ * S_ * D_;          // [B,S,S]

    const size_t QKV = (size_t)B_ * H_ * S_ * D_;          // 16.78M bf16 each
    unsigned short* Qb = (unsigned short*)d_ws;            // [B,H,S,D] bf16
    unsigned short* Kb = Qb + QKV;
    unsigned short* Vb = Kb + QKV;
    unsigned short* Zc = Vb + QKV;                         // [B,S,H*D] bf16

    // zero attn_avg (accumulated via atomics)
    {
        const size_t n = (size_t)B_ * S_ * S_;
        zero_f32<<<dim3((unsigned)((n + 255) / 256)), 256, 0, stream>>>(attn_avg, n);
    }

    const float inv4 = 0.21022410381342865f;               // D^{-1/4}

    // per-(b,h) projections: [S,D] @ [D,D] -> bf16, q/k scaled by D^-1/4
    dim3 gproj(D_ / 64, S_ / 128, B_ * H_);
    gemm_wmma<true, true, true, false><<<gproj, 256, 0, stream>>>(
        Xq, Wq, Qb, nullptr, S_, D_, D_,
        (long long)S_ * D_, H_, (long long)D_ * D_, H_, (long long)S_ * D_, inv4);
    gemm_wmma<true, true, true, false><<<gproj, 256, 0, stream>>>(
        Xk, Wk, Kb, nullptr, S_, D_, D_,
        (long long)S_ * D_, H_, (long long)D_ * D_, H_, (long long)S_ * D_, inv4);
    gemm_wmma<true, true, true, false><<<gproj, 256, 0, stream>>>(
        Xv, Wv, Vb, nullptr, S_, D_, D_,
        (long long)S_ * D_, H_, (long long)D_ * D_, H_, (long long)S_ * D_, 1.0f);

    // attention: 512 workgroups of (b, h, 64 q-rows)
    dim3 gattn(S_ / 64, H_, B_);
    attn_kernel<<<gattn, 256, 0, stream>>>(Qb, Kb, Vb, Zc, attn_avg);

    // output projection: [B*S, H*D] @ [H*D, D] + bz -> f32
    dim3 gout(D_ / 64, (B_ * S_) / 128, 1);
    gemm_wmma<false, true, false, true><<<gout, 256, 0, stream>>>(
        Zc, Wz, out, bz, B_ * S_, D_, HD_, 0, 1, 0, 1, 0, 1.0f);
}
